// BalancedHamiltonLayer_52621939311124
// MI455X (gfx1250) — matrix operand order repaired and checked by reading, not compiled, not run on hardware
//
#include <hip/hip_runtime.h>

typedef __attribute__((ext_vector_type(16))) _Float16 v16h;
typedef __attribute__((ext_vector_type(8)))  float    v8f;

#define DIM   1024
#define NTOK  12288   // 48 * 256

__device__ __forceinline__ v8f wmma16(v16h a, v16h b, v8f c) {
  // D = A(16x32 f16) * B(32x16 f16) + C(16x16 f32)
  return __builtin_amdgcn_wmma_f32_16x16x32_f16(
      /*neg_a=*/false, a, /*neg_b=*/false, b,
      /*c_mod=*/(short)0, c, /*reuse_a=*/false, /*reuse_b=*/false);
}

// ---------------------------------------------------------------------------
// Kernel 1: build W[din][dout] = sum_r H[r,dout/32,din/32] * B[r,dout%32,din%32]
// (H = Hamilton(A_stack), evaluated analytically via sign/component tables),
// split each f32 into f16 hi/lo, and store pre-swizzled in WMMA B-operand
// layout:  element (K=kk in chunk kc, N=nl in tile ct) goes to
//   lane = nl + 16*(kk>=16), half = kk%16
//   off  = ((kc*64 + ct)*32 + lane)*16 + half
// ---------------------------------------------------------------------------
__global__ __launch_bounds__(256) void prep_w(const float* __restrict__ A,
                                              const float* __restrict__ Bs,
                                              _Float16* __restrict__ Whi,
                                              _Float16* __restrict__ Wlo) {
  const int idx  = blockIdx.x * 256 + threadIdx.x;   // [0, 1M)
  const int din  = idx & (DIM - 1);
  const int dout = idx >> 10;

  const int i = din  & 31, s = din  >> 5;
  const int j = dout & 31, k = dout >> 5;
  const int p = k >> 3, a = k & 7;
  const int q = s >> 3, b = s & 7;

  // Hamilton block structure: H[p][q] = sgn[p][q] * A_component[comp[p][q]]
  const int   comp[4][4] = {{0,1,2,3},{1,0,3,2},{2,3,0,1},{3,2,1,0}};
  const float sgn [4][4] = {{ 1.f,-1.f,-1.f,-1.f},
                            { 1.f, 1.f,-1.f, 1.f},
                            { 1.f, 1.f, 1.f,-1.f},
                            { 1.f,-1.f, 1.f, 1.f}};
  const int   cpq = comp[p][q];
  const float sg  = sgn[p][q];

  float w = 0.f;
  #pragma unroll
  for (int r = 0; r < 8; ++r) {
    const float h  = A[((r * 4 + cpq) * 8 + a) * 8 + b];
    const float bb = Bs[(r * 32 + j) * 32 + i];      // B_stack already scaled
    w += h * bb;
  }
  w *= sg;

  const _Float16 hi = (_Float16)w;
  const _Float16 lo = (_Float16)(w - (float)hi);

  const int kc = din >> 5, kk = din & 31;
  const int ct = dout >> 4, nl = dout & 15;
  const int lane = nl + ((kk & 16) ? 16 : 0);
  const int half = kk & 15;
  const size_t off = (((size_t)(kc * 64 + ct)) * 32 + lane) * 16 + half;
  Whi[off] = hi;
  Wlo[off] = lo;
}

// ---------------------------------------------------------------------------
// Kernel 2: out(12288x1024) = x(12288x1024) @ W + bias, split-f16 WMMA GEMM.
// 8 waves/block; wave -> 16 tokens x 64 output cols (4 C tiles).
// ---------------------------------------------------------------------------
__global__ __launch_bounds__(256) void qgemm(const float* __restrict__ x,
                                             const _Float16* __restrict__ Whi,
                                             const _Float16* __restrict__ Wlo,
                                             const float* __restrict__ bias,
                                             float* __restrict__ out) {
  const int lane    = threadIdx.x & 31;
  const int wid     = threadIdx.x >> 5;
  const int ttile   = blockIdx.x * 8 + wid;   // 0..767 (16-token tile)
  const int cgrp    = blockIdx.y;             // 0..15  (64-col group)
  const int nl      = lane & 15;
  const int hiHalf  = lane >> 4;              // 0 or 1
  const int koffA   = hiHalf * 16;            // A-layout lane K offset

  const float* xr = x + (size_t)(ttile * 16 + nl) * DIM;

  v8f c0 = {}, c1 = {}, c2 = {}, c3 = {};
  const float b0 = bias[cgrp * 64 +  0 + nl];
  const float b1 = bias[cgrp * 64 + 16 + nl];
  const float b2 = bias[cgrp * 64 + 32 + nl];
  const float b3 = bias[cgrp * 64 + 48 + nl];

  // Per-lane base of the pre-swizzled W tiles for this column group.
  const size_t wlanebase = (((size_t)(cgrp * 4)) * 32 + lane) * 16;
  const _Float16* wh = Whi + wlanebase;
  const _Float16* wl = Wlo + wlanebase;

  #pragma unroll 1
  for (int kc = 0; kc < 32; ++kc) {
    const int base = kc * 32;
    // A fragment: contiguous K order -> one contiguous 16-float run/lane (k = 16*hiHalf + s)
    const float4 r0a = *(const float4*)(xr + base + koffA);
    const float4 r0b = *(const float4*)(xr + base + koffA + 4);
    const float4 r1a = *(const float4*)(xr + base + koffA + 8);
    const float4 r1b = *(const float4*)(xr + base + koffA + 12);
    if (kc < 31) __builtin_prefetch(xr + base + 32 + koffA, 0, 0);

    v16h ahi, alo;
#define CVT(ix, val)                                   \
    { const float v_ = (val);                          \
      const _Float16 h_ = (_Float16)v_;                \
      ahi[ix] = h_;                                    \
      alo[ix] = (_Float16)(v_ - (float)h_); }
    CVT(0,  r0a.x) CVT(1,  r0a.y) CVT(2,  r0a.z) CVT(3,  r0a.w)
    CVT(4,  r0b.x) CVT(5,  r0b.y) CVT(6,  r0b.z) CVT(7,  r0b.w)
    CVT(8,  r1a.x) CVT(9,  r1a.y) CVT(10, r1a.z) CVT(11, r1a.w)
    CVT(12, r1b.x) CVT(13, r1b.y) CVT(14, r1b.z) CVT(15, r1b.w)
#undef CVT

    const size_t kb = (size_t)kc * (64 * 32 * 16);   // halfs per K-chunk slab
    {
      const v16h bh = *(const v16h*)(wh + kb + 0 * 512);
      const v16h bl = *(const v16h*)(wl + kb + 0 * 512);
      c0 = wmma16(ahi, bh, c0); c0 = wmma16(alo, bh, c0); c0 = wmma16(ahi, bl, c0);
    }
    {
      const v16h bh = *(const v16h*)(wh + kb + 1 * 512);
      const v16h bl = *(const v16h*)(wl + kb + 1 * 512);
      c1 = wmma16(ahi, bh, c1); c1 = wmma16(alo, bh, c1); c1 = wmma16(ahi, bl, c1);
    }
    {
      const v16h bh = *(const v16h*)(wh + kb + 2 * 512);
      const v16h bl = *(const v16h*)(wl + kb + 2 * 512);
      c2 = wmma16(ahi, bh, c2); c2 = wmma16(alo, bh, c2); c2 = wmma16(ahi, bl, c2);
    }
    {
      const v16h bh = *(const v16h*)(wh + kb + 3 * 512);
      const v16h bl = *(const v16h*)(wl + kb + 3 * 512);
      c3 = wmma16(ahi, bh, c3); c3 = wmma16(alo, bh, c3); c3 = wmma16(ahi, bl, c3);
    }
  }

  // C/D layout: VGPR g, lane -> token row = 16*ttile + g + 8*hiHalf, col = nl
  const int mOff = hiHalf * 8;
  #pragma unroll
  for (int g = 0; g < 8; ++g) {
    float* o = out + (size_t)(ttile * 16 + mOff + g) * DIM + cgrp * 64 + nl;
    o[0]  = c0[g] + b0;
    o[16] = c1[g] + b1;
    o[32] = c2[g] + b2;
    o[48] = c3[g] + b3;
  }
}

// ---------------------------------------------------------------------------
extern "C" void kernel_launch(void* const* d_in, const int* in_sizes, int n_in,
                              void* d_out, int out_size, void* d_ws, size_t ws_size,
                              hipStream_t stream) {
  const float* x    = (const float*)d_in[0];   // (48,256,1024)
  const float* A    = (const float*)d_in[1];   // (8,4,8,8)
  const float* Bs   = (const float*)d_in[2];   // (8,32,32) pre-scaled
  const float* bias = (const float*)d_in[3];   // (1024,)
  float* out = (float*)d_out;

  _Float16* Whi = (_Float16*)d_ws;                     // 2 MB
  _Float16* Wlo = Whi + (size_t)DIM * DIM;             // 2 MB

  prep_w<<<(DIM * DIM) / 256, 256, 0, stream>>>(A, Bs, Whi, Wlo);

  dim3 grid(NTOK / 128, DIM / 64);                     // 96 x 16 blocks
  qgemm<<<grid, 256, 0, stream>>>(x, Whi, Wlo, bias, out);
}
